// FeatureConvertor_51591147160090
// MI455X (gfx1250) — matrix operand order repaired and checked by reading, not compiled, not run on hardware
//
#include <hip/hip_runtime.h>

// ---------------------------------------------------------------------------
// MI455X (gfx1250, wave32) implementation of the 4-layer Mamba-ish stack.
// GEMMs: v_wmma_f32_16x16x32_bf16 with LDS tiles filled by async global->LDS.
// ---------------------------------------------------------------------------

typedef __bf16 bf16_t;
typedef __attribute__((ext_vector_type(16))) __bf16 v16bf;
typedef __attribute__((ext_vector_type(8)))  __bf16 v8bf;
typedef __attribute__((ext_vector_type(8)))  float  v8f;
typedef __attribute__((ext_vector_type(4)))  int    v4i;

static constexpr int Bb   = 2;
static constexpr int Lseq = 1024;
static constexpr int DIN  = 512;
static constexpr int Hd   = 768;
static constexpr int DOUTc= 512;
static constexpr int NLAY = 4;
static constexpr int ED   = 768;
static constexpr int Nst  = 16;
static constexpr int DTRc = 48;
static constexpr int DTRP = 64;       // 48 zero-padded to multiple of BK
static constexpr int Mrows= Bb * Lseq; // 2048

#define CAT16(lo, hi) __builtin_shufflevector(lo, hi, 0,1,2,3,4,5,6,7,8,9,10,11,12,13,14,15)

// --- CDNA5 async global->LDS copy (ASYNCcnt-tracked); plain-store fallback ---
#if __has_builtin(__builtin_amdgcn_global_load_async_to_lds_b128)
typedef __attribute__((address_space(1))) v4i* as1_v4i_p;
typedef __attribute__((address_space(3))) v4i* as3_v4i_p;
__device__ __forceinline__ void async_cp16(const bf16_t* g, bf16_t* l)
{
    __builtin_amdgcn_global_load_async_to_lds_b128(
        (as1_v4i_p)(v4i*)(void*)const_cast<bf16_t*>(g),
        (as3_v4i_p)(v4i*)(void*)l, 0, 0);
}
  #if __has_builtin(__builtin_amdgcn_s_wait_asynccnt)
    #define ASYNC_WAIT() __builtin_amdgcn_s_wait_asynccnt(0)
  #else
    #define ASYNC_WAIT() asm volatile("s_wait_asynccnt 0x0" ::: "memory")
  #endif
#else
__device__ __forceinline__ void async_cp16(const bf16_t* g, bf16_t* l)
{
    *(v8bf*)l = *(const v8bf*)g;
}
  #define ASYNC_WAIT()
#endif

// ---------------------------------------------------------------------------
// WMMA GEMM: C[m,n] = act( sum_k A[m,k]*W[n,k] + bias[n] ) (+ resid[m,n])
// A: (M x K) bf16 row-major, lda; W: (N x K) bf16 row-major, ldw.
// Block: 256 thr = 8 waves; block tile 128x128, BK=64 staged in LDS;
// wave tile 32x64 -> 16 WMMAs per BK stage.  ACT: 0 none, 1 relu, 2 softplus.
// Requires: K % 64 == 0, M % 128 == 0 (N handled by clamp+predicated store).
// ---------------------------------------------------------------------------
static constexpr int BM = 128, BN = 128, BK = 64, LSTR = 72; // padded LDS stride

template <int ACT>
__global__ __launch_bounds__(256)
void k_gemm(const bf16_t* __restrict__ A, int lda,
            const bf16_t* __restrict__ W, int ldw,
            const float* __restrict__ bias,
            const float* __restrict__ resid,
            float* __restrict__ Cf,
            bf16_t* __restrict__ Cbf,
            int Ndim, int Kdim)
{
    __shared__ __align__(16) bf16_t smA[BM * LSTR];
    __shared__ __align__(16) bf16_t smB[BN * LSTR];

    const int tid  = threadIdx.x;
    const int lane = tid & 31;
    const int wave = tid >> 5;
    const int r    = lane & 15;   // fragment row (A: M, B/C: N)
    const int hh   = lane >> 4;   // K-half selector
    const int wm   = wave >> 1;   // 0..3
    const int wn   = wave & 1;    // 0..1
    const int m0   = blockIdx.y * BM;
    const int n0   = blockIdx.x * BN;

    v8f acc[2][4] = {};

    for (int kk = 0; kk < Kdim; kk += BK) {
        // ---- cooperative async tile fill: 4 x 16B chunks of A and B each ----
        #pragma unroll
        for (int c = 0; c < 4; ++c) {
            const int chunk = c * 256 + tid;     // 0..1023
            const int row   = chunk >> 3;        // 0..127
            const int cc    = (chunk & 7) * 8;   // K element offset (16B units)
            const bf16_t* ga = A + (size_t)(m0 + row) * lda + kk + cc;
            async_cp16(ga, &smA[row * LSTR + cc]);
            int nrow = n0 + row;
            if (nrow > Ndim - 1) nrow = Ndim - 1;          // stay in-bounds
            const bf16_t* gb = W + (size_t)nrow * ldw + kk + cc;
            async_cp16(gb, &smB[row * LSTR + cc]);
        }
        ASYNC_WAIT();
        __syncthreads();

        // ---- 2 K-steps of 32 from LDS ----
        #pragma unroll
        for (int ks = 0; ks < 2; ++ks) {
            const int kb = ks * 32;
            v16bf a[2], b[4];
            #pragma unroll
            for (int ti = 0; ti < 2; ++ti) {
                const bf16_t* p = &smA[(wm * 32 + ti * 16 + r) * LSTR + kb + 8 * hh];
                v8bf lo = *(const v8bf*)(p);
                v8bf hi = *(const v8bf*)(p + 16);
                a[ti] = CAT16(lo, hi);
            }
            #pragma unroll
            for (int tj = 0; tj < 4; ++tj) {
                const bf16_t* p = &smB[(wn * 64 + tj * 16 + r) * LSTR + kb + 8 * hh];
                v8bf lo = *(const v8bf*)(p);
                v8bf hi = *(const v8bf*)(p + 16);
                b[tj] = CAT16(lo, hi);
            }
            #pragma unroll
            for (int ti = 0; ti < 2; ++ti)
                #pragma unroll
                for (int tj = 0; tj < 4; ++tj)
                    acc[ti][tj] = __builtin_amdgcn_wmma_f32_16x16x32_bf16(
                        false, a[ti], false, b[tj], (short)0, acc[ti][tj], false, false);
        }
        __syncthreads();
    }

    // ---- epilogue: C/D layout -> row = i + 8*hh, col = lane&15 ----
    #pragma unroll
    for (int ti = 0; ti < 2; ++ti) {
        #pragma unroll
        for (int tj = 0; tj < 4; ++tj) {
            const int col = n0 + wn * 64 + tj * 16 + r;
            if (col >= Ndim) continue;
            const int rowb = m0 + wm * 32 + ti * 16 + 8 * hh;
            const float bv = bias ? bias[col] : 0.0f;
            #pragma unroll
            for (int i = 0; i < 8; ++i) {
                float v = acc[ti][tj][i] + bv;
                if (ACT == 1) v = fmaxf(v, 0.0f);
                else if (ACT == 2) v = (v > 20.0f) ? v : __logf(1.0f + __expf(v));
                const size_t off = (size_t)(rowb + i) * Ndim + col;
                if (resid) v += resid[off];
                if (Cf)  Cf[off]  = v;
                if (Cbf) Cbf[off] = (bf16_t)v;
            }
        }
    }
}

// ---------------------------------------------------------------------------
// fp32 -> bf16 convert with optional K zero-padding (dst width ld_dst >= cols)
// ---------------------------------------------------------------------------
__global__ void k_cvt_pad(const float* __restrict__ src, int ld_src,
                          bf16_t* __restrict__ dst, int ld_dst, int cols, long total)
{
    long i = (long)blockIdx.x * blockDim.x + threadIdx.x;
    if (i >= total) return;
    int  c = (int)(i % ld_dst);
    long r = i / ld_dst;
    float v = (c < cols) ? src[r * ld_src + c] : 0.0f;
    dst[i] = (bf16_t)v;
}

// ---------------------------------------------------------------------------
// LayerNorm over last dim (ncol) of a (rows x ncol) fp32 tensor.
// ---------------------------------------------------------------------------
__global__ __launch_bounds__(256)
void k_layernorm(const float* __restrict__ src,
                 const float* __restrict__ w, const float* __restrict__ b,
                 float* __restrict__ outf, bf16_t* __restrict__ outbf, int ncol)
{
    __shared__ float s1[256], s2[256];
    const long  row = blockIdx.x;
    const float* p  = src + row * ncol;
    float s = 0.0f, q = 0.0f;
    for (int c = threadIdx.x; c < ncol; c += 256) { float v = p[c]; s += v; q += v * v; }
    s1[threadIdx.x] = s; s2[threadIdx.x] = q;
    __syncthreads();
    for (int off = 128; off > 0; off >>= 1) {
        if ((int)threadIdx.x < off) {
            s1[threadIdx.x] += s1[threadIdx.x + off];
            s2[threadIdx.x] += s2[threadIdx.x + off];
        }
        __syncthreads();
    }
    const float mean = s1[0] / ncol;
    const float var  = s2[0] / ncol - mean * mean;
    const float inv  = rsqrtf(var + 1e-5f);
    for (int c = threadIdx.x; c < ncol; c += 256) {
        float v = (p[c] - mean) * inv * w[c] + b[c];
        if (outf)  outf[row * ncol + c]  = v;
        if (outbf) outbf[row * ncol + c] = (bf16_t)v;
    }
}

// ---------------------------------------------------------------------------
// RMSNorm over last dim, bf16 output (feeds GEMM only).
// ---------------------------------------------------------------------------
__global__ __launch_bounds__(256)
void k_rmsnorm(const float* __restrict__ src, const float* __restrict__ w,
               bf16_t* __restrict__ outbf, int ncol)
{
    __shared__ float s2[256];
    const long  row = blockIdx.x;
    const float* p  = src + row * ncol;
    float q = 0.0f;
    for (int c = threadIdx.x; c < ncol; c += 256) { float v = p[c]; q += v * v; }
    s2[threadIdx.x] = q;
    __syncthreads();
    for (int off = 128; off > 0; off >>= 1) {
        if ((int)threadIdx.x < off) s2[threadIdx.x] += s2[threadIdx.x + off];
        __syncthreads();
    }
    const float inv = rsqrtf(s2[0] / ncol + 1e-5f);
    for (int c = threadIdx.x; c < ncol; c += 256)
        outbf[row * ncol + c] = (bf16_t)(p[c] * inv * w[c]);
}

// ---------------------------------------------------------------------------
// Depthwise causal conv (K=4) over L + bias + SiLU. Input = xz[:, :ED] (ld 1536).
// ---------------------------------------------------------------------------
__global__ void k_conv_silu(const float* __restrict__ xz,
                            const float* __restrict__ cw, const float* __restrict__ cb,
                            float* __restrict__ xif, bf16_t* __restrict__ xibf)
{
    int idx = blockIdx.x * blockDim.x + threadIdx.x;
    if (idx >= Mrows * ED) return;
    const int e  = idx % ED;
    const int ml = idx / ED;         // b*L + l
    const int l  = ml % Lseq;
    float acc = cb[e];
    #pragma unroll
    for (int j = 0; j < 4; ++j) {
        const int ll = l - 3 + j;
        if (ll >= 0) acc += cw[e * 4 + j] * xz[(size_t)(ml - 3 + j) * (2 * ED) + e];
    }
    const float s = acc / (1.0f + __expf(-acc));
    xif[idx]  = s;
    xibf[idx] = (bf16_t)s;
}

// ---------------------------------------------------------------------------
// Selective scan: per (b,e) register-resident recurrence over L with 16 states.
// ---------------------------------------------------------------------------
__global__ void k_scan(const float* __restrict__ delta, const float* __restrict__ xi,
                       const float* __restrict__ dbc,   const float* __restrict__ xz,
                       const float* __restrict__ Alog,  const float* __restrict__ Dp,
                       bf16_t* __restrict__ ybf)
{
    const int t = blockIdx.x * blockDim.x + threadIdx.x;
    if (t >= Bb * ED) return;
    const int e = t % ED;
    const int b = t / ED;

    float Arow[Nst];
    #pragma unroll
    for (int n = 0; n < Nst; ++n) Arow[n] = -__expf(Alog[e * Nst + n]);
    const float dpe = Dp[e];

    float hs[Nst] = {};
    for (int l = 0; l < Lseq; ++l) {
        const size_t idx = (size_t)(b * Lseq + l);
        const float dl = delta[idx * ED + e];
        const float xv = xi[idx * ED + e];
        const float dx = dl * xv;
        float Bm[Nst], Cm[Nst];
        const float* pr = dbc + idx * (DTRc + 2 * Nst);
        #pragma unroll
        for (int q = 0; q < 4; ++q) {
            ((float4*)Bm)[q] = ((const float4*)(pr + DTRc))[q];
            ((float4*)Cm)[q] = ((const float4*)(pr + DTRc + Nst))[q];
        }
        float y = 0.0f;
        #pragma unroll
        for (int n = 0; n < Nst; ++n) {
            hs[n] = __expf(dl * Arow[n]) * hs[n] + dx * Bm[n];
            y += hs[n] * Cm[n];
        }
        y += dpe * xv;
        const float z = xz[idx * (2 * ED) + ED + e];
        y *= z / (1.0f + __expf(-z));
        ybf[idx * ED + e] = (bf16_t)y;
    }
}

// ---------------------------------------------------------------------------
// h = h + xln + t2
// ---------------------------------------------------------------------------
__global__ void k_add3(float* __restrict__ h, const float* __restrict__ xln,
                       const float* __restrict__ t2, long n)
{
    long i = (long)blockIdx.x * blockDim.x + threadIdx.x;
    if (i < n) h[i] = h[i] + xln[i] + t2[i];
}

// ---------------------------------------------------------------------------
// Host-side launchers
// ---------------------------------------------------------------------------
static inline void cvt(const float* src, int lds, bf16_t* dst, int ldd, int cols,
                       long rows, hipStream_t s)
{
    long total = rows * (long)ldd;
    int blocks = (int)((total + 255) / 256);
    k_cvt_pad<<<blocks, 256, 0, s>>>(src, lds, dst, ldd, cols, total);
}

static inline void gemm(const bf16_t* A, int lda, const bf16_t* W, int ldw,
                        const float* bias, const float* resid,
                        float* Cf, bf16_t* Cbf, int Mm, int Nn, int Kk, int act,
                        hipStream_t s)
{
    dim3 g((Nn + BN - 1) / BN, Mm / BM);
    if (act == 0)
        k_gemm<0><<<g, 256, 0, s>>>(A, lda, W, ldw, bias, resid, Cf, Cbf, Nn, Kk);
    else if (act == 1)
        k_gemm<1><<<g, 256, 0, s>>>(A, lda, W, ldw, bias, resid, Cf, Cbf, Nn, Kk);
    else
        k_gemm<2><<<g, 256, 0, s>>>(A, lda, W, ldw, bias, resid, Cf, Cbf, Nn, Kk);
}

extern "C" void kernel_launch(void* const* d_in, const int* in_sizes, int n_in,
                              void* d_out, int out_size, void* d_ws, size_t ws_size,
                              hipStream_t stream)
{
    (void)in_sizes; (void)n_in; (void)out_size; (void)ws_size;

    const float* x         = (const float*)d_in[0];
    const float* in_proj_w = (const float*)d_in[1];
    const float* in_proj_b = (const float*)d_in[2];
    const float* out_proj_w= (const float*)d_in[3];
    const float* out_proj_b= (const float*)d_in[4];
    const float* ln1_w     = (const float*)d_in[5];
    const float* ln1_b     = (const float*)d_in[6];
    const float* ln2_w     = (const float*)d_in[7];
    const float* ln2_b     = (const float*)d_in[8];
    const float* ffn_w1    = (const float*)d_in[9];
    const float* ffn_b1    = (const float*)d_in[10];
    const float* ffn_w2    = (const float*)d_in[11];
    const float* ffn_b2    = (const float*)d_in[12];
    const float* rms_w     = (const float*)d_in[13];
    const float* m_in_w    = (const float*)d_in[14];
    const float* conv_w    = (const float*)d_in[15];
    const float* conv_b    = (const float*)d_in[16];
    const float* xproj_w   = (const float*)d_in[17];
    const float* dt_w      = (const float*)d_in[18];
    const float* dt_b      = (const float*)d_in[19];
    const float* A_log     = (const float*)d_in[20];
    const float* Dp        = (const float*)d_in[21];
    const float* m_out_w   = (const float*)d_in[22];
    float* out = (float*)d_out;

    // workspace carve-out (256B aligned)
    char*  ws  = (char*)d_ws;
    size_t off = 0;
    auto alloc = [&](size_t bytes) -> void* {
        off = (off + 255) & ~(size_t)255;
        void* p = ws + off;
        off += bytes;
        return p;
    };
    float* h     = (float*)alloc((size_t)Mrows * Hd * 4);
    float* xln   = (float*)alloc((size_t)Mrows * Hd * 4);
    float* xz    = (float*)alloc((size_t)Mrows * 2 * ED * 4);
    float* xif   = (float*)alloc((size_t)Mrows * ED * 4);
    float* dbc   = (float*)alloc((size_t)Mrows * (DTRc + 2 * Nst) * 4);
    float* delta = (float*)alloc((size_t)Mrows * ED * 4);
    float* t2    = (float*)alloc((size_t)Mrows * Hd * 4);

    bf16_t* xbf  = (bf16_t*)alloc((size_t)Mrows * DIN * 2);
    bf16_t* xnbf = (bf16_t*)alloc((size_t)Mrows * ED * 2);
    bf16_t* xibf = (bf16_t*)alloc((size_t)Mrows * ED * 2);
    bf16_t* dbf  = (bf16_t*)alloc((size_t)Mrows * DTRP * 2);
    bf16_t* ybf  = (bf16_t*)alloc((size_t)Mrows * ED * 2);
    bf16_t* hnbf = (bf16_t*)alloc((size_t)Mrows * Hd * 2);
    bf16_t* ubf  = (bf16_t*)alloc((size_t)Mrows * 4 * Hd * 2);
    bf16_t* hbf  = (bf16_t*)alloc((size_t)Mrows * Hd * 2);
    bf16_t* wbf  = (bf16_t*)alloc((size_t)(4 * Hd) * Hd * 2); // largest weight

    const int ew_blocks = (Mrows * ED + 255) / 256;

    // ----- h = x @ in_proj_w.T + in_proj_b -----
    cvt(x, DIN, xbf, DIN, DIN, Mrows, stream);
    cvt(in_proj_w, DIN, wbf, DIN, DIN, Hd, stream);
    gemm(xbf, DIN, wbf, DIN, in_proj_b, nullptr, h, nullptr, Mrows, Hd, DIN, 0, stream);

    for (int i = 0; i < NLAY; ++i) {
        // xln = LN1(h)
        k_layernorm<<<Mrows, 256, 0, stream>>>(h, ln1_w + i * Hd, ln1_b + i * Hd,
                                               xln, nullptr, Hd);
        // xn = RMSNorm(xln) -> bf16
        k_rmsnorm<<<Mrows, 256, 0, stream>>>(xln, rms_w + i * ED, xnbf, ED);
        // xz = xn @ m_in_w.T (2048 x 1536)
        cvt(m_in_w + (size_t)i * 2 * ED * Hd, Hd, wbf, Hd, Hd, 2 * ED, stream);
        gemm(xnbf, Hd, wbf, Hd, nullptr, nullptr, xz, nullptr, Mrows, 2 * ED, Hd, 0, stream);
        // xi = silu(causal_conv(xz[:, :ED]) + cb)
        k_conv_silu<<<ew_blocks, 256, 0, stream>>>(xz, conv_w + (size_t)i * ED * 4,
                                                   conv_b + i * ED, xif, xibf);
        // dbc = xi @ xproj_w.T (2048 x 80)
        cvt(xproj_w + (size_t)i * (DTRc + 2 * Nst) * ED, ED, wbf, ED, ED,
            DTRc + 2 * Nst, stream);
        gemm(xibf, ED, wbf, ED, nullptr, nullptr, dbc, nullptr,
             Mrows, DTRc + 2 * Nst, ED, 0, stream);
        // delta = softplus(d @ dt_w.T + dt_b) with K padded 48->64
        cvt(dbc, DTRc + 2 * Nst, dbf, DTRP, DTRc, Mrows, stream);
        cvt(dt_w + (size_t)i * ED * DTRc, DTRc, wbf, DTRP, DTRc, ED, stream);
        gemm(dbf, DTRP, wbf, DTRP, dt_b + i * ED, nullptr, delta, nullptr,
             Mrows, ED, DTRP, 2, stream);
        // selective scan -> y (bf16)
        k_scan<<<(Bb * ED + 255) / 256, 256, 0, stream>>>(
            delta, xif, dbc, xz, A_log + (size_t)i * ED * Nst, Dp + i * ED, ybf);
        // t2 = y @ m_out_w.T
        cvt(m_out_w + (size_t)i * Hd * ED, ED, wbf, ED, ED, Hd, stream);
        gemm(ybf, ED, wbf, ED, nullptr, nullptr, t2, nullptr, Mrows, Hd, ED, 0, stream);
        // h = h + xln + t2
        k_add3<<<ew_blocks, 256, 0, stream>>>(h, xln, t2, (long)Mrows * Hd);
        // FFN
        k_layernorm<<<Mrows, 256, 0, stream>>>(h, ln2_w + i * Hd, ln2_b + i * Hd,
                                               nullptr, hnbf, Hd);
        cvt(ffn_w1 + (size_t)i * 4 * Hd * Hd, Hd, wbf, Hd, Hd, 4 * Hd, stream);
        gemm(hnbf, Hd, wbf, Hd, ffn_b1 + (size_t)i * 4 * Hd, nullptr,
             nullptr, ubf, Mrows, 4 * Hd, Hd, 1, stream);
        cvt(ffn_w2 + (size_t)i * Hd * 4 * Hd, 4 * Hd, wbf, 4 * Hd, 4 * Hd, Hd, stream);
        gemm(ubf, 4 * Hd, wbf, 4 * Hd, ffn_b2 + i * Hd, h,
             h, nullptr, Mrows, Hd, 4 * Hd, 0, stream);
    }

    // ----- out = h @ out_proj_w.T + out_proj_b -----
    cvt(h, Hd, hbf, Hd, Hd, Mrows, stream);
    cvt(out_proj_w, Hd, wbf, Hd, Hd, DOUTc, stream);
    gemm(hbf, Hd, wbf, Hd, out_proj_b, nullptr, out, nullptr, Mrows, DOUTc, Hd, 0, stream);
}
